// BigBirdProteinModel_15857019257171
// MI455X (gfx1250) — compile-verified, matrix operand order checked by reading, not run on hardware
//
#include <hip/hip_runtime.h>
#include <hip/hip_bf16.h>
#include <math.h>

#define SEQL 4096
#define NBLK 64
#define NH   12
#define HD   64
#define DMO  768
#define DFF  3072
#define NL   12
#define BSZ  2
#define MROWS (BSZ*SEQL)   // 8192
#define TGT  500
#define INDIM 1280

typedef __attribute__((ext_vector_type(16))) __bf16 v16bf;
typedef __attribute__((ext_vector_type(8)))  __bf16 v8bf;
typedef __attribute__((ext_vector_type(8)))  float  v8f;
typedef __attribute__((ext_vector_type(4)))  float  f32x4;
typedef __attribute__((ext_vector_type(4)))  unsigned int u32x4;
typedef __attribute__((ext_vector_type(8)))  int i32x8;
typedef __attribute__((ext_vector_type(4)))  int i32x4;

#define HAVE_TDM __has_builtin(__builtin_amdgcn_tensor_load_to_lds)

// ---------------------------------------------------------------------------
// WMMA: D(16x16 f32) = A(16x32 bf16) * B(32x16 bf16) + C   (wave32)
// ---------------------------------------------------------------------------
__device__ inline v8f wmma_bf16(v16bf a, v16bf b, v8f c) {
  return __builtin_amdgcn_wmma_f32_16x16x32_bf16(
      false, a, false, b, (short)0, c, false, false);
}

// Fragment from a row-major [rows][ld] bf16 LDS tile (A operand, or B operand
// stored as [n][k]).  ISA 16-bit 16x32 layout: lanes 0-15 -> K base 0,
// lanes 16-31 -> K base 8; elems 0..7 -> kb+0..7, elems 8..15 -> kb+16..23.
// ld and koff chosen so every access is a 16B-aligned v8bf -> ds_load_b128.
__device__ inline v16bf frag_ld(const __bf16* p, int row0, int ld, int koff,
                                int lane) {
  int m  = row0 + (lane & 15);
  int kb = koff + ((lane & 16) ? 8 : 0);
  const v8bf* q = (const v8bf*)(p + m * ld + kb);
  v8bf lo = q[0];
  v8bf hi = q[2];  // +16 elements
  return __builtin_shufflevector(lo, hi, 0, 1, 2, 3, 4, 5, 6, 7,
                                 8, 9, 10, 11, 12, 13, 14, 15);
}

__device__ inline float gelu_tanh(float x) {
  float x3 = x * x * x;
  return 0.5f * x * (1.0f + tanhf(0.7978845608028654f * (x + 0.044715f * x3)));
}

// ---------------------------------------------------------------------------
// Tensor Data Mover: 2-D tile 64 rows x 64 bf16 (128B/row), row stride
// `stride_elems`, into LDS at byte offset lds_off with 8-elem row padding
// (pad 4 dwords every 32 dwords) -> LDS row stride 72 elems, matching frag_ld.
// ---------------------------------------------------------------------------
#if HAVE_TDM
__device__ inline void tdm_load_tile64(unsigned lds_off, const __bf16* g,
                                       unsigned stride_elems) {
  unsigned long long ga = (unsigned long long)(uintptr_t)g;
  u32x4 g0;
  g0[0] = 1u;                                   // count=1, user descriptor
  g0[1] = lds_off;                              // LDS byte address
  g0[2] = (unsigned)ga;                         // global_addr[31:0]
  g0[3] = ((unsigned)(ga >> 32) & 0x1FFFFFFu)   // global_addr[56:32]
        | (2u << 30);                           // type = 2 (image)
  i32x8 g1;
  g1[0] = (int)((1u << 16)      // data_size = 2B
              | (1u << 20)      // pad_enable
              | (4u << 22)      // pad_interval: every 32 dwords
              | (3u << 25));    // pad_amount: 4 dwords
  g1[1] = (int)(64u << 16);     // tensor_dim0 = 64 (bits 79:48, low16)
  g1[2] = (int)(64u << 16);     // tensor_dim0 hi=0 | tensor_dim1 low16 = 64
  g1[3] = (int)(64u << 16);     // tensor_dim1 hi=0 | tile_dim0 = 64
  g1[4] = (int)64u;             // tile_dim1 = 64, tile_dim2 = 0
  g1[5] = (int)stride_elems;    // tensor_dim0_stride low32 (data_size units)
  g1[6] = 0;                    // stride hi16 | tensor_dim1_stride low16
  g1[7] = 0;
  i32x4 z4 = {0, 0, 0, 0};
  i32x8 z8 = {0, 0, 0, 0, 0, 0, 0, 0};
  __builtin_amdgcn_tensor_load_to_lds(g0, g1, z4, z4, z8, 0);
}
__device__ inline unsigned lds_byte_off(const void* p) {
  // generic pointers to LDS: addr[31:0] is the LDS byte offset (ISA 10.2)
  return (unsigned)(uintptr_t)p;
}
#endif

// ---------------------------------------------------------------------------
// GEMM: C[M,N] = act((A[M,K] @ B[K,N] + bias[N]) * scale)
// A,B fp32 in HBM -> bf16 in LDS; BsT staged transposed so both operands are
// contiguous row-major for ds_load_b128 fragments.  Software-pipelined:
// tile kt+1 is prefetched into registers while tile kt's WMMAs execute.
// OUT: 0 = fp32 [M][N], 1 = bf16 [M][N], 2 = bf16 per-head transposed
//      [b][h][hd][seq] (for V; enables raw TDM tiles in attention).
// Requires M%64==0, N%128==0, K%32==0 (holds for all call sites).
// ---------------------------------------------------------------------------
template <int ACT, int OUT>  // ACT: 0 none, 1 gelu
__global__ __launch_bounds__(256) void gemm_bf16wmma(
    const float* __restrict__ A, const float* __restrict__ B,
    const float* __restrict__ bias, void* __restrict__ Cv,
    int M, int N, int K, float scale) {
  constexpr int BM = 64, BN = 128, BK = 32, LDK = 40;  // 40: pad vs bank conflicts
  __shared__ __bf16 As[BM][LDK];    // 5 KB
  __shared__ __bf16 BsT[BN][LDK];   // 10 KB, [n][k]
  const int bm = blockIdx.y * BM, bn = blockIdx.x * BN;
  const int tid = threadIdx.x, wid = tid >> 5, lane = tid & 31;
  const int wrow = (wid >> 2) * 32, wcol = (wid & 3) * 32;
  // staging assignments (contiguous per-thread runs, coalesced across threads)
  const int am = tid >> 2, ach = (tid & 3) * 8;      // A: row, 8-elem chunk
  const int bnl = tid & 127, bkh = (tid >> 7) * 16;  // B: col, 16-row half
  const float* Abase = A + (size_t)(bm + am) * K + ach;
  const float* Bbase = B + (size_t)bkh * N + bn + bnl;

  v8f acc[2][2];
#pragma unroll
  for (int i = 0; i < 2; ++i)
#pragma unroll
    for (int j = 0; j < 2; ++j)
#pragma unroll
      for (int r = 0; r < 8; ++r) acc[i][j][r] = 0.0f;

  // prefetch tile kt=0 into registers
  f32x4 pa0 = *(const f32x4*)(Abase + 0);
  f32x4 pa1 = *(const f32x4*)(Abase + 4);
  float pb[16];
#pragma unroll
  for (int i = 0; i < 16; ++i) pb[i] = Bbase[(size_t)i * N];

  for (int kt = 0; kt < K; kt += BK) {
    __syncthreads();  // previous tile's consumers done
    // publish prefetched tile to LDS (bf16)
    v8bf apk;
#pragma unroll
    for (int i = 0; i < 4; ++i) { apk[i] = (__bf16)pa0[i]; apk[4 + i] = (__bf16)pa1[i]; }
    *(v8bf*)&As[am][ach] = apk;
    v8bf bp0, bp1;
#pragma unroll
    for (int i = 0; i < 8; ++i) { bp0[i] = (__bf16)pb[i]; bp1[i] = (__bf16)pb[8 + i]; }
    *(v8bf*)&BsT[bnl][bkh] = bp0;
    *(v8bf*)&BsT[bnl][bkh + 8] = bp1;
    __syncthreads();

    // issue next tile's global loads early; latency hides behind WMMAs below
    if (kt + BK < K) {
      pa0 = *(const f32x4*)(Abase + kt + BK);
      pa1 = *(const f32x4*)(Abase + kt + BK + 4);
      const float* bs = Bbase + (size_t)(kt + BK) * N;
#pragma unroll
      for (int i = 0; i < 16; ++i) pb[i] = bs[(size_t)i * N];
    }

    v16bf a0 = frag_ld(&As[0][0], wrow, LDK, 0, lane);
    v16bf a1 = frag_ld(&As[0][0], wrow + 16, LDK, 0, lane);
    v16bf b0 = frag_ld(&BsT[0][0], wcol, LDK, 0, lane);
    v16bf b1 = frag_ld(&BsT[0][0], wcol + 16, LDK, 0, lane);
    acc[0][0] = wmma_bf16(a0, b0, acc[0][0]);
    acc[0][1] = wmma_bf16(a0, b1, acc[0][1]);
    acc[1][0] = wmma_bf16(a1, b0, acc[1][0]);
    acc[1][1] = wmma_bf16(a1, b1, acc[1][1]);
  }

  const int m0 = (lane & 16) ? 8 : 0;
  const int nn = lane & 15;
#pragma unroll
  for (int i = 0; i < 2; ++i)
#pragma unroll
    for (int j = 0; j < 2; ++j) {
      int col = bn + wcol + 16 * j + nn;
      int row0 = bm + wrow + 16 * i + m0;
      float bia = bias[col];
      if (OUT == 0) {
        float* C = (float*)Cv;
#pragma unroll
        for (int r = 0; r < 8; ++r) {
          float v = (acc[i][j][r] + bia) * scale;
          if (ACT == 1) v = gelu_tanh(v);
          C[(size_t)(row0 + r) * N + col] = v;
        }
      } else if (OUT == 1) {
        __bf16* C = (__bf16*)Cv;
#pragma unroll
        for (int r = 0; r < 8; ++r) {
          float v = (acc[i][j][r] + bia) * scale;
          C[(size_t)(row0 + r) * N + col] = (__bf16)v;
        }
      } else {
        __bf16* C = (__bf16*)Cv;  // [b][h][hd][seq]; 8 consecutive seq / lane
        int bb = row0 / SEQL, s0 = row0 % SEQL;
        int hh = col / HD, hd = col % HD;
        v8bf pk;
#pragma unroll
        for (int r = 0; r < 8; ++r) pk[r] = (__bf16)((acc[i][j][r] + bia) * scale);
        *(v8bf*)(C + (((size_t)bb * NH + hh) * HD + hd) * SEQL + s0) = pk;
      }
    }
}

// ---------------------------------------------------------------------------
// BigBird attention, flash-style online softmax over the per-query-block key
// block list.  grid=(NBLK, NH, BSZ), block=128 (4 waves, 16 q-rows each).
// Q,K bf16 [b][s][dm] (Q pre-scaled by 1/8); V bf16 transposed [b][h][hd][s].
// K/V tiles are double-buffered and pulled asynchronously by the Tensor Data
// Mover: tile j+1 streams in while tile j's WMMAs run (TDM ops complete in
// order per wave, so s_wait_tensorcnt(2) guarantees the current pair landed).
// ---------------------------------------------------------------------------
#define ALD 72  // padded LDS row stride (elems): +4 dwords per 32 dwords
__global__ __launch_bounds__(128) void bigbird_attn(
    const __bf16* __restrict__ Q, const __bf16* __restrict__ K,
    const __bf16* __restrict__ VT, const float* __restrict__ maskf,
    const int* __restrict__ idxtab, const int* __restrict__ nkbtab,
    const float* __restrict__ duptab, float* __restrict__ O) {
  __shared__ __bf16 Qs[64][ALD];
  __shared__ __bf16 Ps[64][ALD];
  __shared__ __bf16 Ks[2][64][ALD];
  __shared__ __bf16 Vs[2][64][ALD];  // [hd][key]

  const int qb = blockIdx.x, hh = blockIdx.y, b = blockIdx.z;
  const int tid = threadIdx.x, wid = tid >> 5, lane = tid & 31;
  const int m0 = (lane & 16) ? 8 : 0;
  const int nn = lane & 15;

  const __bf16* qg = Q + ((size_t)b * SEQL + qb * 64) * DMO + hh * HD;
  const __bf16* Kbase = K + (size_t)b * SEQL * DMO + hh * HD;
  const __bf16* Vbase = VT + ((size_t)b * NH + hh) * HD * SEQL;
  const int nkb = nkbtab[qb];

#if HAVE_TDM
  if (wid == 0) {  // prologue: Q + first K/V pair in flight
    int kb0 = idxtab[qb * 64];
    tdm_load_tile64(lds_byte_off(&Qs[0][0]), qg, DMO);
    tdm_load_tile64(lds_byte_off(&Ks[0][0][0]), Kbase + (size_t)kb0 * 64 * DMO, DMO);
    tdm_load_tile64(lds_byte_off(&Vs[0][0][0]), Vbase + (size_t)kb0 * 64, SEQL);
  }
#else
  for (int e = tid; e < 512; e += 128) {
    int r = e >> 3, ch = (e & 7) * 8;
    *(v8bf*)&Qs[r][ch] = *(const v8bf*)(qg + (size_t)r * DMO + ch);
    int kb0 = idxtab[qb * 64];
    *(v8bf*)&Ks[0][r][ch] =
        *(const v8bf*)(Kbase + ((size_t)kb0 * 64 + r) * DMO + ch);
    *(v8bf*)&Vs[0][r][ch] =
        *(const v8bf*)(Vbase + (size_t)kb0 * 64 + (size_t)r * SEQL + ch);
  }
#endif

  float rowM[8], rowL[8];
#pragma unroll
  for (int r = 0; r < 8; ++r) { rowM[r] = -INFINITY; rowL[r] = 0.0f; }
  v8f accO[4];
#pragma unroll
  for (int t = 0; t < 4; ++t)
#pragma unroll
    for (int r = 0; r < 8; ++r) accO[t][r] = 0.0f;

  for (int j = 0; j < nkb; ++j) {
    const int buf = j & 1;
    const int kb = idxtab[qb * 64 + j];
    const float dupv = duptab[qb * 64 + j];
    __syncthreads();  // all waves done reading buf^1 (iteration j-1)
#if HAVE_TDM
    if (wid == 0) {
      if (j + 1 < nkb) {  // prefetch next pair into the other buffer
        int kbn = idxtab[qb * 64 + j + 1];
        tdm_load_tile64(lds_byte_off(&Ks[buf ^ 1][0][0]),
                        Kbase + (size_t)kbn * 64 * DMO, DMO);
        tdm_load_tile64(lds_byte_off(&Vs[buf ^ 1][0][0]),
                        Vbase + (size_t)kbn * 64, SEQL);
        __builtin_amdgcn_s_wait_tensorcnt(2);  // current pair resident
      } else {
        __builtin_amdgcn_s_wait_tensorcnt(0);
      }
    }
#else
    if (j + 1 < nkb) {
      int kbn = idxtab[qb * 64 + j + 1];
      for (int e = tid; e < 512; e += 128) {
        int r = e >> 3, ch = (e & 7) * 8;
        *(v8bf*)&Ks[buf ^ 1][r][ch] =
            *(const v8bf*)(Kbase + ((size_t)kbn * 64 + r) * DMO + ch);
        *(v8bf*)&Vs[buf ^ 1][r][ch] =
            *(const v8bf*)(Vbase + (size_t)kbn * 64 + (size_t)r * SEQL + ch);
      }
    }
#endif
    __syncthreads();
    const __bf16* Kt = &Ks[buf][0][0];
    const __bf16* Vt = &Vs[buf][0][0];

    // S(16x64) = Q_w @ K_blk^T; K^T as [n=key][k=hd] == K rows -> frag_ld
    v8f s[4];
#pragma unroll
    for (int t = 0; t < 4; ++t)
#pragma unroll
      for (int r = 0; r < 8; ++r) s[t][r] = 0.0f;
#pragma unroll
    for (int kk = 0; kk < 64; kk += 32) {
      v16bf aq = frag_ld(&Qs[0][0], wid * 16, ALD, kk, lane);
#pragma unroll
      for (int t = 0; t < 4; ++t) {
        v16bf bk = frag_ld(Kt, t * 16, ALD, kk, lane);
        s[t] = wmma_bf16(aq, bk, s[t]);
      }
    }

    // mask (key mask * dup mask): masked -> -1e9
#pragma unroll
    for (int t = 0; t < 4; ++t) {
      int key = kb * 64 + t * 16 + nn;
      float mv = maskf[(size_t)b * SEQL + key] * dupv;
      if (mv <= 0.0f) {
#pragma unroll
        for (int r = 0; r < 8; ++r) s[t][r] = -1.0e9f;
      }
    }

    // online softmax per row (reduce across the 16 lanes of each half-wave)
#pragma unroll
    for (int r = 0; r < 8; ++r) {
      float mx = s[0][r];
#pragma unroll
      for (int t = 1; t < 4; ++t) mx = fmaxf(mx, s[t][r]);
      for (int d = 1; d < 16; d <<= 1) mx = fmaxf(mx, __shfl_xor(mx, d, 32));
      float nm = fmaxf(rowM[r], mx);
      float corr = (rowM[r] == -INFINITY) ? 0.0f : __expf(rowM[r] - nm);
      float sum = 0.0f;
#pragma unroll
      for (int t = 0; t < 4; ++t) {
        float p = __expf(s[t][r] - nm);
        s[t][r] = p;
        sum += p;
      }
      for (int d = 1; d < 16; d <<= 1) sum += __shfl_xor(sum, d, 32);
      rowL[r] = rowL[r] * corr + sum;
      rowM[r] = nm;
#pragma unroll
      for (int t = 0; t < 4; ++t) accO[t][r] *= corr;
    }

    // P -> per-wave LDS slab (C-layout scatter; same-wave consumer)
#pragma unroll
    for (int t = 0; t < 4; ++t)
#pragma unroll
      for (int r = 0; r < 8; ++r)
        Ps[wid * 16 + m0 + r][t * 16 + nn] = (__bf16)s[t][r];

    // O += P(16x64) @ V_blk; B as [n=hd][k=key] == Vs rows -> frag_ld
#pragma unroll
    for (int kk = 0; kk < 64; kk += 32) {
      v16bf ap = frag_ld(&Ps[0][0], wid * 16, ALD, kk, lane);
#pragma unroll
      for (int t = 0; t < 4; ++t) {
        v16bf bv = frag_ld(Vt, t * 16, ALD, kk, lane);
        accO[t] = wmma_bf16(ap, bv, accO[t]);
      }
    }
  }

  float* Obase = O + ((size_t)b * SEQL + qb * 64 + wid * 16) * DMO + hh * HD;
#pragma unroll
  for (int t = 0; t < 4; ++t)
#pragma unroll
    for (int r = 0; r < 8; ++r) {
      float l = rowL[r];
      float v = (l > 0.0f) ? accO[t][r] / l : 0.0f;
      Obase[(size_t)(m0 + r) * DMO + t * 16 + nn] = v;
    }
}

// ---------------------------------------------------------------------------
// LayerNorm with optional residual: out = LN(x (+ res)) * g + b  (eps 1e-12)
// ---------------------------------------------------------------------------
__global__ __launch_bounds__(256) void ln_kernel(
    const float* __restrict__ x, const float* __restrict__ res,
    const float* __restrict__ g, const float* __restrict__ bta,
    float* __restrict__ out, int n) {
  __shared__ float r1[256], r2[256];
  const int row = blockIdx.x, tid = threadIdx.x;
  const float* xr = x + (size_t)row * n;
  const float* rr = res ? res + (size_t)row * n : nullptr;
  float s = 0.0f, s2 = 0.0f;
  for (int c = tid; c < n; c += 256) {
    float v = xr[c] + (rr ? rr[c] : 0.0f);
    s += v; s2 += v * v;
  }
  r1[tid] = s; r2[tid] = s2;
  __syncthreads();
  for (int st = 128; st > 0; st >>= 1) {
    if (tid < st) { r1[tid] += r1[tid + st]; r2[tid] += r2[tid + st]; }
    __syncthreads();
  }
  float mu = r1[0] / n;
  float var = r2[0] / n - mu * mu;
  float inv = rsqrtf(var + 1e-12f);
  float* orow = out + (size_t)row * n;
  for (int c = tid; c < n; c += 256) {
    float v = xr[c] + (rr ? rr[c] : 0.0f);
    orow[c] = (v - mu) * inv * g[c] + bta[c];
  }
}

__global__ void add_emb(float* __restrict__ h, const float* __restrict__ pos,
                        const float* __restrict__ typ) {
  size_t i = (size_t)blockIdx.x * blockDim.x + threadIdx.x;
  if (i >= (size_t)MROWS * DMO) return;
  int c = (int)(i % DMO);
  int sidx = (int)((i / DMO) % SEQL);
  h[i] += pos[(size_t)sidx * DMO + c] + typ[c];
}

__global__ void mask_cast(const int* __restrict__ am, float* __restrict__ mf, int n) {
  int i = blockIdx.x * blockDim.x + threadIdx.x;
  if (i < n) mf[i] = (float)am[i];
}

// Key-block tables. Rows 0/63 -> global (all 64 blocks). Rows 1..62 -> base
// window {0,63,i-1,i,i+1} + 3 distinct pseudo-random extras; dup-mask zeroes
// repeated blocks (first occurrence wins), matching the reference structure.
__global__ void init_tabs(int* __restrict__ idxtab, int* __restrict__ nkbtab,
                          float* __restrict__ duptab) {
  int i = threadIdx.x;
  if (i >= NBLK) return;
  if (i == 0 || i == NBLK - 1) {
    nkbtab[i] = 64;
    for (int j = 0; j < 64; ++j) { idxtab[i * 64 + j] = j; duptab[i * 64 + j] = 1.0f; }
  } else {
    nkbtab[i] = 8;
    int row[8];
    row[0] = 0; row[1] = NBLK - 1; row[2] = i - 1; row[3] = i; row[4] = i + 1;
    unsigned st = 0x9E3779B9u * (unsigned)i + 12345u;
    int cnt = 5;
    while (cnt < 8) {
      st = st * 1664525u + 1013904223u;
      int c = (int)((st >> 16) & 63);
      bool dup = false;
      for (int t = 0; t < cnt; ++t) if (row[t] == c) dup = true;
      if (!dup) row[cnt++] = c;
    }
    for (int j = 0; j < 8; ++j) {
      float m = 1.0f;
      for (int t = 0; t < j; ++t) if (row[t] == row[j]) m = 0.0f;
      idxtab[i * 64 + j] = row[j];
      duptab[i * 64 + j] = m;
    }
    for (int j = 8; j < 64; ++j) { idxtab[i * 64 + j] = 0; duptab[i * 64 + j] = 0.0f; }
  }
}

// Tiny classifier GEMMs (M = 2 rows): scalar is fine.
__global__ void small_fc(const float* __restrict__ in, size_t instride,
                         const float* __restrict__ W, const float* __restrict__ bias,
                         float* __restrict__ out, int K, int N, int act) {
  int b = blockIdx.y;
  int n = blockIdx.x * blockDim.x + threadIdx.x;
  if (n >= N) return;
  const float* ir = in + (size_t)b * instride;
  float acc = bias[n];
  for (int k = 0; k < K; ++k) acc += ir[k] * W[(size_t)k * N + n];
  if (act == 1) acc = fmaxf(acc, 0.0f);
  out[(size_t)b * N + n] = acc;
}

// ---------------------------------------------------------------------------
extern "C" void kernel_launch(void* const* d_in, const int* in_sizes, int n_in,
                              void* d_out, int out_size, void* d_ws, size_t ws_size,
                              hipStream_t stream) {
  (void)in_sizes; (void)n_in; (void)out_size; (void)ws_size;
  const float* x      = (const float*)d_in[0];
  const int*   amask  = (const int*)d_in[1];
  const float* proj_w = (const float*)d_in[2];
  const float* proj_b = (const float*)d_in[3];
  const float* pos    = (const float*)d_in[4];
  const float* typ    = (const float*)d_in[5];
  const float* emb_g  = (const float*)d_in[6];
  const float* emb_b  = (const float*)d_in[7];
  const float* Wq = (const float*)d_in[8];
  const float* bq = (const float*)d_in[9];
  const float* Wk = (const float*)d_in[10];
  const float* bk = (const float*)d_in[11];
  const float* Wv = (const float*)d_in[12];
  const float* bv = (const float*)d_in[13];
  const float* Wo = (const float*)d_in[14];
  const float* bo = (const float*)d_in[15];
  const float* g1 = (const float*)d_in[16];
  const float* be1 = (const float*)d_in[17];
  const float* W1 = (const float*)d_in[18];
  const float* b1 = (const float*)d_in[19];
  const float* W2 = (const float*)d_in[20];
  const float* b2 = (const float*)d_in[21];
  const float* g2 = (const float*)d_in[22];
  const float* be2 = (const float*)d_in[23];
  const float* c1w = (const float*)d_in[24];
  const float* c1b = (const float*)d_in[25];
  const float* c2w = (const float*)d_in[26];
  const float* c2b = (const float*)d_in[27];
  const float* c3w = (const float*)d_in[28];
  const float* c3b = (const float*)d_in[29];
  float* yout = (float*)d_out;

  // workspace layout (byte allocator, 64B aligned)
  char* base = (char*)d_ws;
  size_t off = 0;
  auto alloc = [&](size_t bytes) {
    void* p = base + off;
    off += (bytes + 63) & ~(size_t)63;
    return p;
  };
  float* h     = (float*)alloc((size_t)MROWS * DMO * 4);
  float* h1    = (float*)alloc((size_t)MROWS * DMO * 4);
  float* aobuf = (float*)alloc((size_t)MROWS * DMO * 4);
  float* tbuf  = (float*)alloc((size_t)MROWS * DMO * 4);
  float* ffmid = (float*)alloc((size_t)MROWS * DFF * 4);
  __bf16* qb16 = (__bf16*)alloc((size_t)MROWS * DMO * 2);
  __bf16* kb16 = (__bf16*)alloc((size_t)MROWS * DMO * 2);
  __bf16* vbT  = (__bf16*)alloc((size_t)MROWS * DMO * 2);
  float* clsa  = (float*)alloc((size_t)BSZ * DMO * 4);
  float* clsb  = (float*)alloc((size_t)BSZ * 512 * 4);
  float* maskf = (float*)alloc((size_t)BSZ * SEQL * 4);
  float* dupt  = (float*)alloc((size_t)NBLK * 64 * 4);
  int*   idxt  = (int*)alloc((size_t)NBLK * 64 * 4);
  int*   nkbt  = (int*)alloc((size_t)NBLK * 4);

  init_tabs<<<1, 64, 0, stream>>>(idxt, nkbt, dupt);
  mask_cast<<<(BSZ * SEQL + 255) / 256, 256, 0, stream>>>(amask, maskf, BSZ * SEQL);

  const dim3 blk256(256), blk128(128);
  const dim3 gDM(DMO / 128, MROWS / 64);
  const dim3 gFF(DFF / 128, MROWS / 64);

  // embedding: x @ proj_w + proj_b ; + pos + type ; LN
  gemm_bf16wmma<0, 0><<<gDM, blk256, 0, stream>>>(x, proj_w, proj_b, h,
                                                  MROWS, DMO, INDIM, 1.0f);
  add_emb<<<((size_t)MROWS * DMO + 255) / 256, blk256, 0, stream>>>(h, pos, typ);
  ln_kernel<<<MROWS, blk256, 0, stream>>>(h, nullptr, emb_g, emb_b, h, DMO);

  for (int l = 0; l < NL; ++l) {
    const float* wq = Wq + (size_t)l * DMO * DMO;
    const float* wk = Wk + (size_t)l * DMO * DMO;
    const float* wv = Wv + (size_t)l * DMO * DMO;
    const float* wo = Wo + (size_t)l * DMO * DMO;
    const float* w1 = W1 + (size_t)l * DMO * DFF;
    const float* w2 = W2 + (size_t)l * DFF * DMO;
    const float* bql = bq + (size_t)l * DMO;
    const float* bkl = bk + (size_t)l * DMO;
    const float* bvl = bv + (size_t)l * DMO;
    const float* bol = bo + (size_t)l * DMO;
    const float* b1l = b1 + (size_t)l * DFF;
    const float* b2l = b2 + (size_t)l * DMO;
    const float* g1l = g1 + (size_t)l * DMO;
    const float* be1l = be1 + (size_t)l * DMO;
    const float* g2l = g2 + (size_t)l * DMO;
    const float* be2l = be2 + (size_t)l * DMO;

    // QKV projections -> bf16 (Q pre-scaled by 1/sqrt(HD)); V transposed/head
    gemm_bf16wmma<0, 1><<<gDM, blk256, 0, stream>>>(h, wq, bql, qb16,
                                                    MROWS, DMO, DMO, 0.125f);
    gemm_bf16wmma<0, 1><<<gDM, blk256, 0, stream>>>(h, wk, bkl, kb16,
                                                    MROWS, DMO, DMO, 1.0f);
    gemm_bf16wmma<0, 2><<<gDM, blk256, 0, stream>>>(h, wv, bvl, vbT,
                                                    MROWS, DMO, DMO, 1.0f);

    bigbird_attn<<<dim3(NBLK, NH, BSZ), blk128, 0, stream>>>(
        qb16, kb16, vbT, maskf, idxt, nkbt, dupt, aobuf);

    gemm_bf16wmma<0, 0><<<gDM, blk256, 0, stream>>>(aobuf, wo, bol, tbuf,
                                                    MROWS, DMO, DMO, 1.0f);
    ln_kernel<<<MROWS, blk256, 0, stream>>>(tbuf, h, g1l, be1l, h1, DMO);

    gemm_bf16wmma<1, 0><<<gFF, blk256, 0, stream>>>(h1, w1, b1l, ffmid,
                                                    MROWS, DFF, DMO, 1.0f);
    gemm_bf16wmma<0, 0><<<gDM, blk256, 0, stream>>>(ffmid, w2, b2l, tbuf,
                                                    MROWS, DMO, DFF, 1.0f);
    ln_kernel<<<MROWS, blk256, 0, stream>>>(tbuf, h1, g2l, be2l, h, DMO);
  }

  // classifier head on CLS token (row 0 of each batch)
  small_fc<<<dim3((DMO + 255) / 256, BSZ), blk256, 0, stream>>>(
      h, (size_t)SEQL * DMO, c1w, c1b, clsa, DMO, DMO, 1);
  small_fc<<<dim3((512 + 255) / 256, BSZ), blk256, 0, stream>>>(
      clsa, (size_t)DMO, c2w, c2b, clsb, DMO, 512, 1);
  small_fc<<<dim3((TGT + 255) / 256, BSZ), blk256, 0, stream>>>(
      clsb, (size_t)512, c3w, c3b, yout, 512, TGT, 0);
}
